// StackedAttentionNetwork_25426206392952
// MI455X (gfx1250) — matrix operand order repaired and checked by reading, not compiled, not run on hardware
//
#include <hip/hip_runtime.h>
#include <math.h>

// ---------------------------------------------------------------------------
// StackedAttentionNetwork for MI455X (gfx1250, wave32, WMMA)
// GEMMs in bf16 WMMA (f32 accumulate) with ping-pong double-buffered K loop
// (no register rotation); everything else fp32 VALU.
// ---------------------------------------------------------------------------

typedef __bf16 bf16;
typedef __attribute__((ext_vector_type(16))) __bf16 v16bf;
typedef __attribute__((ext_vector_type(8)))  __bf16 v8bf;
typedef __attribute__((ext_vector_type(8)))  float  v8f;

#define DI __device__ __forceinline__

static constexpr int B_    = 128;
static constexpr int T_    = 64;
static constexpr int E_    = 300;
static constexpr int EP_   = 320;          // E padded to /32
static constexpr int C_    = 512;
static constexpr int NAREA = 196;          // 14*14
static constexpr int HID_  = 1024;
static constexpr int G4_   = 4096;         // 4*HID
static constexpr int K_    = 512;
static constexpr int MTXT  = T_ * B_;      // 8192 rows of x_flat
static constexpr int MIMG  = B_ * NAREA;   // 25088 rows of img_r

// ---------------------------------------------------------------------------
// WMMA fragment loaders.  C = A(MxK) * B(NxK)^T, A/B row-major bf16, K%32==0.
// A 16x32 bf16 fragment: lane L holds row M=L&15;
//   elems[0..15] = K = (L>>4)*16 + 0..15 (contig)
// B 32x16 fragment: lane L holds col N=L&15, K = (L>>4)*16 + 0..15 (contig).
// ---------------------------------------------------------------------------
DI v16bf load_a_frag(const bf16* __restrict__ A, int lda, int m0, int k0, int lane) {
  int r = lane & 15, kh = lane >> 4;
  return *(const v16bf*)(A + (size_t)(m0 + r) * lda + (k0 + kh * 16));
}

DI v16bf load_b_frag(const bf16* __restrict__ Bm, int ldb, int n0, int k0, int lane) {
  int n = lane & 15, kh = lane >> 4;
  return *(const v16bf*)(Bm + (size_t)(n0 + n) * ldb + (k0 + kh * 16));
}

DI void load_frags(const bf16* __restrict__ A, const bf16* __restrict__ Bm,
                   int K, int m0, int nb, int k0, int lane,
                   v16bf& a, v16bf b[4]) {
  a = load_a_frag(A, K, m0, k0, lane);
#pragma unroll
  for (int j = 0; j < 4; ++j) b[j] = load_b_frag(Bm, K, nb + j * 16, k0, lane);
}

DI void wmma4(const v16bf& a, const v16bf b[4], v8f acc[4]) {
#pragma unroll
  for (int j = 0; j < 4; ++j)
    acc[j] = __builtin_amdgcn_wmma_f32_16x16x32_bf16(
        false, a, false, b[j], (short)0, acc[j], false, false);
}

// One K-segment of acc += A * B^T for a 16x64 wave tile.
// Ping-pong double buffer, K loop unrolled by 2: compute from set 0 while
// set 1 loads, then compute from set 1 while set 0 reloads -- no register
// rotation moves, loads always in flight under the WMMAs.
DI void gemm_segment(const bf16* __restrict__ A, const bf16* __restrict__ Bm,
                     int K, int m0, int nb, int lane, v8f acc[4]) {
  v16bf a0, b0[4], a1, b1[4];
  load_frags(A, Bm, K, m0, nb, 0, lane, a0, b0);

  int k0 = 0;
  while (k0 + 32 < K) {
    if (k0 + 96 < K) {  // global_prefetch_b8 one block beyond the window
      __builtin_prefetch(A + (size_t)(m0 + (lane & 15)) * K + k0 + 96, 0, 1);
      __builtin_prefetch(Bm + (size_t)(nb + lane) * K + k0 + 96, 0, 1);
    }
    load_frags(A, Bm, K, m0, nb, k0 + 32, lane, a1, b1);
    wmma4(a0, b0, acc);
    if (k0 + 64 < K) load_frags(A, Bm, K, m0, nb, k0 + 64, lane, a0, b0);
    wmma4(a1, b1, acc);
    k0 += 64;
  }
  if (k0 < K) wmma4(a0, b0, acc);  // tail (odd step count)
}

// ---------------------------------------------------------------------------
// GEMM: Cf[M,N] = A0*B0^T (+ A1*B1^T) (+ bias[col]); optional bf16 copy.
// Block = 8 waves (32x8), covers 32 rows x 256 cols; wave tile = 16x64.
// grid = (N/256, M/32).  lda/ldb = K of the segment (rows are K-padded).
// ---------------------------------------------------------------------------
__global__ __launch_bounds__(256)
void gemm_bf16_wmma(const bf16* __restrict__ A0, const bf16* __restrict__ B0, int K0,
                    const bf16* __restrict__ A1, const bf16* __restrict__ B1, int K1,
                    const float* __restrict__ bias,
                    float* __restrict__ Cf, bf16* __restrict__ Cbf, int ldc) {
  const int lane = threadIdx.x;
  const int wave = threadIdx.y;
  const int m0 = blockIdx.y * 32 + (wave >> 2) * 16;
  const int nb = blockIdx.x * 256 + (wave & 3) * 64;

  v8f acc[4] = {};
  gemm_segment(A0, B0, K0, m0, nb, lane, acc);
  if (A1 != nullptr)  // fused second K-segment (h @ W_hh^T)
    gemm_segment(A1, B1, K1, m0, nb, lane, acc);

  // C/D layout: VGPR jj holds row M = (lane>>4)*8 + jj, col N = lane&15.
  const int rh = lane >> 4, cl = lane & 15;
#pragma unroll
  for (int j = 0; j < 4; ++j) {
    int col = nb + j * 16 + cl;
    float bv = bias ? bias[col] : 0.0f;
#pragma unroll
    for (int jj = 0; jj < 8; ++jj) {
      int row = m0 + rh * 8 + jj;
      float v = acc[j][jj] + bv;
      Cf[(size_t)row * ldc + col] = v;
      if (Cbf) Cbf[(size_t)row * ldc + col] = (bf16)v;
    }
  }
}

// ---------------------------------------------------------------------------
// Elementwise / prep kernels
// ---------------------------------------------------------------------------
__global__ void gather_embed_kernel(const int* __restrict__ desc,
                                    const float* __restrict__ emb,
                                    bf16* __restrict__ X) {
  // x = emb[desc].reshape(T,B,E): x_flat row r == emb[desc.flat[r]]
  long idx = (long)blockIdx.x * blockDim.x + threadIdx.x;
  if (idx >= (long)MTXT * EP_) return;
  long r = idx / EP_;
  int  k = (int)(idx % EP_);
  float v = 0.0f;
  if (k < E_) v = emb[(size_t)desc[r] * E_ + k];
  X[idx] = (bf16)v;
}

__global__ void f32_to_bf16_pad_kernel(const float* __restrict__ src,
                                       bf16* __restrict__ dst,
                                       int Kin, int Kp, long total) {
  long idx = (long)blockIdx.x * blockDim.x + threadIdx.x;
  if (idx >= total) return;
  long r = idx / Kp;
  int  k = (int)(idx % Kp);
  float v = (k < Kin) ? src[r * (size_t)Kin + k] : 0.0f;
  dst[idx] = (bf16)v;
}

__global__ void bias_sum_kernel(const float* a, const float* b, float* o) {
  int i = blockIdx.x * blockDim.x + threadIdx.x;
  if (i < G4_) o[i] = a[i] + b[i];
}

__global__ void init_state_kernel(float* c, float* h, bf16* hbf) {
  int i = blockIdx.x * blockDim.x + threadIdx.x;
  if (i < B_ * HID_) { c[i] = 0.0f; h[i] = 0.0f; hbf[i] = (bf16)0.0f; }
}

DI float sigf(float x) { return 1.0f / (1.0f + __expf(-x)); }

__global__ void lstm_pointwise_kernel(const float* __restrict__ G,
                                      float* __restrict__ c,
                                      float* __restrict__ h,
                                      bf16* __restrict__ hbf) {
  int idx = blockIdx.x * blockDim.x + threadIdx.x;
  if (idx >= B_ * HID_) return;
  int b = idx >> 10, j = idx & (HID_ - 1);
  const float* g = G + (size_t)b * G4_;
  float gi = sigf(g[j]);
  float gf = sigf(g[HID_ + j]);
  float gg = tanhf(g[2 * HID_ + j]);
  float go = sigf(g[3 * HID_ + j]);
  float cc = gf * c[idx] + gi * gg;
  float hh = go * tanhf(cc);
  c[idx] = cc;
  h[idx] = hh;
  hbf[idx] = (bf16)hh;
}

// scores[r] = sum_k tanh(t_img[r,k] + t_u[b,k]) * wl_p[layer,k] + bl_p[layer]
__global__ __launch_bounds__(256)
void attn_score_kernel(const float* __restrict__ t_img,
                       const float* __restrict__ t_u,
                       const float* __restrict__ wl_p,
                       const float* __restrict__ bl_p,
                       int layer, float* __restrict__ s) {
  int lane = threadIdx.x;               // 32
  int r = blockIdx.x * 8 + threadIdx.y; // 8 waves/block, row per wave
  if (r >= MIMG) return;
  int b = r / NAREA;
  const float* wp = wl_p + (size_t)layer * K_;
  float acc = 0.0f;
  for (int k = lane; k < K_; k += 32)
    acc += tanhf(t_img[(size_t)r * K_ + k] + t_u[(size_t)b * K_ + k]) * wp[k];
#pragma unroll
  for (int off = 16; off > 0; off >>= 1) acc += __shfl_xor(acc, off, 32);
  if (lane == 0) s[r] = acc + bl_p[layer];
}

// per-batch softmax over 196 areas + faithful-reshape weighted sum:
// v[b,h] = sum_n out_img_flat[b, h*196+n] * p[b,n];  u_out = u_in + v
__global__ __launch_bounds__(256)
void softmax_attend_kernel(const float* __restrict__ s,
                           const float* __restrict__ outimg,
                           const float* __restrict__ u_in,
                           float* __restrict__ u_out,
                           bf16* __restrict__ u_bf) {
  __shared__ float red[256];
  __shared__ float p[NAREA];
  int b = blockIdx.x, tid = threadIdx.x;

  float v = (tid < NAREA) ? s[(size_t)b * NAREA + tid] : -INFINITY;
  red[tid] = v;
  __syncthreads();
  for (int off = 128; off > 0; off >>= 1) {
    if (tid < off) red[tid] = fmaxf(red[tid], red[tid + off]);
    __syncthreads();
  }
  float m = red[0];
  __syncthreads();
  float e = (tid < NAREA) ? __expf(v - m) : 0.0f;
  red[tid] = e;
  __syncthreads();
  for (int off = 128; off > 0; off >>= 1) {
    if (tid < off) red[tid] += red[tid + off];
    __syncthreads();
  }
  float inv = 1.0f / red[0];
  if (tid < NAREA) p[tid] = e * inv;
  __syncthreads();

  const float* base = outimg + (size_t)b * NAREA * HID_;
  for (int h = tid; h < HID_; h += 256) {
    float acc = 0.0f;
    const float* row = base + (size_t)h * NAREA;  // faithful reshape [B,HID,N]
    for (int n = 0; n < NAREA; ++n) acc += row[n] * p[n];
    float un = u_in[(size_t)b * HID_ + h] + acc;
    u_out[(size_t)b * HID_ + h] = un;
    if (u_bf) u_bf[(size_t)b * HID_ + h] = (bf16)un;
  }
}

// ---------------------------------------------------------------------------
// Host orchestration
// ---------------------------------------------------------------------------
static inline int cdiv(long a, long b) { return (int)((a + b - 1) / b); }

extern "C" void kernel_launch(void* const* d_in, const int* in_sizes, int n_in,
                              void* d_out, int out_size, void* d_ws, size_t ws_size,
                              hipStream_t stream) {
  const int*   desc  = (const int*)d_in[0];
  const float* img   = (const float*)d_in[1];
  const float* emb   = (const float*)d_in[2];
  const float* W_ih  = (const float*)d_in[3];
  const float* W_hh  = (const float*)d_in[4];
  const float* b_ih  = (const float*)d_in[5];
  const float* b_hh  = (const float*)d_in[6];
  const float* W_img = (const float*)d_in[7];
  const float* b_img = (const float*)d_in[8];
  const float* Wl_img= (const float*)d_in[9];
  const float* Wl_q  = (const float*)d_in[10];
  const float* bl_q  = (const float*)d_in[11];
  const float* wl_p  = (const float*)d_in[12];
  const float* bl_p  = (const float*)d_in[13];
  float* out = (float*)d_out;

  // workspace carving (256B aligned)
  char* w = (char*)d_ws;
  auto alloc = [&](size_t bytes) {
    char* p = w;
    w += (bytes + 255) & ~(size_t)255;
    return p;
  };
  bf16*  Xbf     = (bf16*)alloc((size_t)MTXT * EP_ * 2);
  bf16*  Wih_bf  = (bf16*)alloc((size_t)G4_ * EP_ * 2);
  bf16*  Whh_bf  = (bf16*)alloc((size_t)G4_ * HID_ * 2);
  float* bsum    = (float*)alloc((size_t)G4_ * 4);
  float* Gbuf    = (float*)alloc((size_t)B_ * G4_ * 4);
  float* cbuf    = (float*)alloc((size_t)B_ * HID_ * 4);
  float* hbuf    = (float*)alloc((size_t)B_ * HID_ * 4);   // u_k lives here
  bf16*  hbf     = (bf16*)alloc((size_t)B_ * HID_ * 2);
  bf16*  img_bf  = (bf16*)alloc((size_t)MIMG * C_ * 2);
  bf16*  Wimg_bf = (bf16*)alloc((size_t)HID_ * C_ * 2);
  float* outimg  = (float*)alloc((size_t)MIMG * HID_ * 4);
  bf16*  outimg_bf = (bf16*)alloc((size_t)MIMG * HID_ * 2);
  bf16*  Wlimg_bf  = (bf16*)alloc((size_t)2 * K_ * HID_ * 2);
  bf16*  Wlq_bf    = (bf16*)alloc((size_t)2 * K_ * HID_ * 2);
  float* t_img   = (float*)alloc((size_t)MIMG * K_ * 4);
  float* t_u     = (float*)alloc((size_t)B_ * K_ * 4);
  float* scores  = (float*)alloc((size_t)MIMG * 4);

  const dim3 blk1(256), blkG(32, 8);

  // --- prep: gathers / bf16 conversions ---
  gather_embed_kernel<<<cdiv((long)MTXT * EP_, 256), blk1, 0, stream>>>(desc, emb, Xbf);
  f32_to_bf16_pad_kernel<<<cdiv((long)G4_ * EP_, 256), blk1, 0, stream>>>(
      W_ih, Wih_bf, E_, EP_, (long)G4_ * EP_);
  f32_to_bf16_pad_kernel<<<cdiv((long)G4_ * HID_, 256), blk1, 0, stream>>>(
      W_hh, Whh_bf, HID_, HID_, (long)G4_ * HID_);
  f32_to_bf16_pad_kernel<<<cdiv((long)MIMG * C_, 256), blk1, 0, stream>>>(
      img, img_bf, C_, C_, (long)MIMG * C_);
  f32_to_bf16_pad_kernel<<<cdiv((long)HID_ * C_, 256), blk1, 0, stream>>>(
      W_img, Wimg_bf, C_, C_, (long)HID_ * C_);
  f32_to_bf16_pad_kernel<<<cdiv((long)2 * K_ * HID_, 256), blk1, 0, stream>>>(
      Wl_img, Wlimg_bf, HID_, HID_, (long)2 * K_ * HID_);
  f32_to_bf16_pad_kernel<<<cdiv((long)2 * K_ * HID_, 256), blk1, 0, stream>>>(
      Wl_q, Wlq_bf, HID_, HID_, (long)2 * K_ * HID_);
  bias_sum_kernel<<<cdiv(G4_, 256), blk1, 0, stream>>>(b_ih, b_hh, bsum);
  init_state_kernel<<<cdiv(B_ * HID_, 256), blk1, 0, stream>>>(cbuf, hbuf, hbf);

  // --- image branch: out_img = img_r @ W_img^T + b_img  (f32 + bf16 copies) ---
  {
    dim3 grid(HID_ / 256, MIMG / 32);
    gemm_bf16_wmma<<<grid, blkG, 0, stream>>>(img_bf, Wimg_bf, C_,
                                              nullptr, nullptr, 0,
                                              b_img, outimg, outimg_bf, HID_);
  }

  // --- LSTM: 64 sequential steps, fused x@W_ih^T + h@W_hh^T per step ---
  {
    dim3 grid(G4_ / 256, B_ / 32);
    for (int t = 0; t < T_; ++t) {
      gemm_bf16_wmma<<<grid, blkG, 0, stream>>>(
          Xbf + (size_t)t * B_ * EP_, Wih_bf, EP_,
          hbf, Whh_bf, HID_,
          bsum, Gbuf, nullptr, G4_);
      lstm_pointwise_kernel<<<cdiv(B_ * HID_, 256), blk1, 0, stream>>>(
          Gbuf, cbuf, hbuf, hbf);
    }
  }

  // --- stacked attention (L = 2) ---
  for (int layer = 0; layer < 2; ++layer) {
    {
      dim3 grid(K_ / 256, MIMG / 32);
      gemm_bf16_wmma<<<grid, blkG, 0, stream>>>(
          outimg_bf, Wlimg_bf + (size_t)layer * K_ * HID_, HID_,
          nullptr, nullptr, 0,
          nullptr, t_img, nullptr, K_);
    }
    {
      dim3 grid(K_ / 256, B_ / 32);
      gemm_bf16_wmma<<<grid, blkG, 0, stream>>>(
          hbf, Wlq_bf + (size_t)layer * K_ * HID_, HID_,
          nullptr, nullptr, 0,
          bl_q + (size_t)layer * K_, t_u, nullptr, K_);
    }
    attn_score_kernel<<<MIMG / 8, blkG, 0, stream>>>(
        t_img, t_u, wl_p, bl_p, layer, scores);
    softmax_attend_kernel<<<B_, blk1, 0, stream>>>(
        scores, outimg, hbuf,
        (layer == 1) ? out : hbuf,
        (layer == 0) ? hbf : nullptr);
  }
}
